// EncoderLayer_20633022890490
// MI455X (gfx1250) — matrix operand order repaired and checked by reading, not compiled, not run on hardware
//
#include <hip/hip_runtime.h>
#include <hip/hip_bf16.h>
#include <math.h>
#include <stdint.h>

// ---------------------------------------------------------------------------
// Autoformer encoder layer for MI455X (gfx1250, wave32).
// B=16, L=4096, D=512 fp32.
//   - autocorrelation scores as per-batch Gram S*S^T on v_wmma_f32_16x16x32_bf16,
//     diagonals binned to tau via ds_add_f32 + global_atomic_add_f32.
//   - FFN GEMMs on WMMA bf16 with hi/lo split fp32 emulation; weight tiles
//     staged to LDS by the Tensor Data Mover (tensor_load_to_lds + TENSORcnt).
//   - moving-average decomp, roll-combine, layernorms fused streaming kernels.
// ---------------------------------------------------------------------------

typedef __attribute__((ext_vector_type(16))) __bf16 v16bf;
typedef __attribute__((ext_vector_type(8)))  float  v8f;
typedef __attribute__((ext_vector_type(4)))  unsigned int tdm_v4u;
typedef __attribute__((ext_vector_type(8)))  int          tdm_v8i;
typedef __attribute__((ext_vector_type(4)))  int          tdm_v4i;

#define BATCH 16
#define SEQL  4096
#define DIM   512
#define TOPK  5

__device__ inline v8f wmma_bf16(v16bf a, v16bf b, v8f c) {
  // (neg_a, A, neg_b, B, c_mod, C, reuse_a, reuse_b)
  return __builtin_amdgcn_wmma_f32_16x16x32_bf16(false, a, false, b, (short)0, c, false, false);
}

// ---------------------------------------------------------------------------
// Tensor Data Mover: async 2-D bf16 tile load global -> LDS.
// Builds D# per CDNA5 ISA 8.3/8.4: group0 {count, lds_addr, global_addr, type=2},
// group1 {data_size=2B, tensor_dim0=row_stride, tensor_dim1=1<<20, tile WxH,
// tensor_dim0_stride=row_stride}. Tile is written packed into LDS.
// Issue from ONE wave only; complete with s_wait_tensorcnt 0.
// This toolchain exposes the 6-arg builtin: (g0 v4u32, g1 v8i32, g2 v4i32,
// g3 v4i32, v8i32, cpol).
// ---------------------------------------------------------------------------
__device__ inline void tdm_load_2d_bf16(const void* gptr, unsigned lds_off,
                                        int tile_w, int tile_h, int row_stride) {
  unsigned long long ga = (unsigned long long)(size_t)gptr;
  tdm_v4u g0;
  g0[0] = 1u;                                               // count=1, user D#
  g0[1] = lds_off;                                          // lds_addr (bytes)
  g0[2] = (unsigned)ga;                                     // global_addr[31:0]
  g0[3] = (unsigned)((ga >> 32) & 0x1ffffffull) | (2u << 30); // addr[56:32] | type=2
  tdm_v8i g1;
  g1[0] = (int)(1u << 16);                                  // data_size=1 -> 2 bytes
  g1[1] = (int)(((unsigned)row_stride & 0xffffu) << 16);    // tensor_dim0[15:0]
  g1[2] = (int)((((unsigned)row_stride >> 16) & 0xffffu));  // dim0 hi | dim1 lo16(=0)
  g1[3] = (int)(16u | ((unsigned)tile_w << 16));            // dim1 hi16 (1<<20) | tile_dim0
  g1[4] = (int)(unsigned)tile_h;                            // tile_dim1 | tile_dim2=0
  g1[5] = row_stride;                                       // tensor_dim0_stride[31:0]
  g1[6] = 0;
  g1[7] = 0;
  tdm_v4i gz4 = {0, 0, 0, 0};
  tdm_v8i gz8 = {0, 0, 0, 0, 0, 0, 0, 0};
  __builtin_amdgcn_tensor_load_to_lds(g0, g1, gz4, gz4, gz8, 0);
}

// A-matrix 16x32 bf16 fragment from row-major LDS (stride in elements).
__device__ inline v16bf load_frag_a(const __bf16* base, int stride, int lane) {
  int row = lane & 15, kg = lane >> 4;
  const __bf16* r = base + row * stride + (kg << 4);
  v16bf f;
#pragma unroll
  for (int v = 0; v < 8; ++v) { f[2*v] = r[2*v]; f[2*v+1] = r[2*v+1]; }
  return f;
}

// B-matrix 32x16 bf16 fragment from LDS stored [k][n] row-major.
__device__ inline v16bf load_frag_b(const __bf16* base, int stride, int lane, int ntile) {
  int col = (lane & 15) + ntile * 16, kg = lane >> 4;
  const __bf16* c = base + ((kg << 4) * stride) + col;
  v16bf f;
#pragma unroll
  for (int v = 0; v < 8; ++v) { f[2*v] = c[(2*v)*stride]; f[2*v+1] = c[(2*v+1)*stride]; }
  return f;
}

// B fragment where LDS holds the tile TRANSPOSED: rows are N, cols are K.
__device__ inline v16bf load_frag_bt(const __bf16* base, int stride, int lane, int ntile) {
  int row = (lane & 15) + ntile * 16, kg = lane >> 4;
  const __bf16* r = base + row * stride + (kg << 4);
  v16bf f;
#pragma unroll
  for (int v = 0; v < 8; ++v) { f[2*v] = r[2*v]; f[2*v+1] = r[2*v+1]; }
  return f;
}

// ---------------------------------------------------------------------------
// Kernel 1: series decomposition (25-tap edge-padded moving average).
// ---------------------------------------------------------------------------
__global__ __launch_bounds__(256) void decomp_kernel(
    const float* __restrict__ X, float* __restrict__ seasonal,
    float* __restrict__ trend_out, const float* __restrict__ trend_add) {
  const int L = SEQL, D = DIM, HW = 12;  // k=25 -> pl=pr=12
  int b = blockIdx.y;
  int l0 = blockIdx.x * 128;
  const float* Xb = X + (size_t)b * L * D;
  for (int dd = 0; dd < 2; ++dd) {
    int d = threadIdx.x + dd * 256;
    float sum = 0.f;
#pragma unroll
    for (int j = -HW; j <= HW; ++j) {
      int l = l0 + j; l = min(max(l, 0), L - 1);
      sum += Xb[(size_t)l * D + d];
    }
    for (int li = 0; li < 128; ++li) {
      int l = l0 + li;
      float tr = sum * (1.0f / 25.0f);
      size_t o = (size_t)b * L * D + (size_t)l * D + d;
      seasonal[o] = Xb[(size_t)l * D + d] - tr;
      trend_out[o] = trend_add ? (tr + trend_add[o]) : tr;
      int lin  = min(l + HW + 1, L - 1);
      int lout = max(l - HW, 0);
      sum += Xb[(size_t)lin * D + d] - Xb[(size_t)lout * D + d];
    }
  }
}

__global__ void zero_kernel(float* p, int n) {
  int i = blockIdx.x * 256 + threadIdx.x;
  if (i < n) p[i] = 0.f;
}

// ---------------------------------------------------------------------------
// Kernel 3: Gram autocorrelation scores on WMMA.
// ---------------------------------------------------------------------------
__global__ __launch_bounds__(128) void gram_scores_kernel(
    const float* __restrict__ S, float* __restrict__ scores) {
  const int L = SEQL, D = DIM;
  __shared__ __bf16 Abf[64][36];
  __shared__ __bf16 Bbf[64][36];
  __shared__ float  taubin[128];
  int b = blockIdx.z;
  int i0 = blockIdx.x * 64, j0 = blockIdx.y * 64;
  const float* Sb = S + (size_t)b * L * D;
  int t = threadIdx.x, lane = t & 31, w = t >> 5;
  v8f acc[4] = {};
  for (int k0 = 0; k0 < D; k0 += 32) {
    for (int e = t; e < 64 * 8; e += 128) {       // 64 rows x 8 float4
      int r = e >> 3, c4 = (e & 7) << 2;
      float4 xa = *(const float4*)(&Sb[(size_t)(i0 + r) * D + k0 + c4]);
      float4 xb = *(const float4*)(&Sb[(size_t)(j0 + r) * D + k0 + c4]);
      Abf[r][c4+0] = (__bf16)xa.x; Abf[r][c4+1] = (__bf16)xa.y;
      Abf[r][c4+2] = (__bf16)xa.z; Abf[r][c4+3] = (__bf16)xa.w;
      Bbf[r][c4+0] = (__bf16)xb.x; Bbf[r][c4+1] = (__bf16)xb.y;
      Bbf[r][c4+2] = (__bf16)xb.z; Bbf[r][c4+3] = (__bf16)xb.w;
    }
    __syncthreads();
    v16bf a = load_frag_a(&Abf[w * 16][0], 36, lane);
#pragma unroll
    for (int nt = 0; nt < 4; ++nt) {
      v16bf bt = load_frag_bt(&Bbf[0][0], 36, lane, nt);
      acc[nt] = wmma_bf16(a, bt, acc[nt]);
    }
    __syncthreads();
  }
  taubin[t] = 0.f;
  __syncthreads();
  int kg = lane >> 4, colb = lane & 15;
#pragma unroll
  for (int nt = 0; nt < 4; ++nt) {
#pragma unroll
    for (int v = 0; v < 8; ++v) {
      int di = w * 16 + v + 8 * kg;
      int dj = nt * 16 + colb;
      atomicAdd(&taubin[dj - di + 63], acc[nt][v]);   // ds_add_f32
    }
  }
  __syncthreads();
  if (t < 127) {
    int tau = ((j0 - i0) + (t - 63)) & (L - 1);
    unsafeAtomicAdd(&scores[(size_t)b * L + tau], taubin[t]);  // global_atomic_add_f32
  }
}

// ---------------------------------------------------------------------------
// Kernel 4: top-5 + softmax (scores scaled by 1/D), one block per batch.
// ---------------------------------------------------------------------------
__global__ __launch_bounds__(256) void topk_softmax_kernel(
    const float* __restrict__ scores, float* __restrict__ wts, int* __restrict__ idxs) {
  const int L = SEQL;
  int b = blockIdx.x;
  __shared__ float sv[256]; __shared__ int si[256];
  __shared__ int   chosen[TOPK]; __shared__ float chval[TOPK];
  const float* sc = scores + (size_t)b * L;
  for (int r = 0; r < TOPK; ++r) {
    float bv = -3.4e38f; int bi = 0;
    for (int l = threadIdx.x; l < L; l += 256) {
      bool skip = false;
      for (int q = 0; q < r; ++q) skip |= (chosen[q] == l);
      if (skip) continue;
      float v = sc[l];
      if (v > bv || (v == bv && l < bi)) { bv = v; bi = l; }
    }
    sv[threadIdx.x] = bv; si[threadIdx.x] = bi;
    __syncthreads();
    for (int s = 128; s > 0; s >>= 1) {
      if (threadIdx.x < s) {
        float ov = sv[threadIdx.x + s]; int oi = si[threadIdx.x + s];
        if (ov > sv[threadIdx.x] || (ov == sv[threadIdx.x] && oi < si[threadIdx.x])) {
          sv[threadIdx.x] = ov; si[threadIdx.x] = oi;
        }
      }
      __syncthreads();
    }
    if (threadIdx.x == 0) { chosen[r] = si[0]; chval[r] = sv[0]; }
    __syncthreads();
  }
  if (threadIdx.x == 0) {
    float m = -3.4e38f;
    for (int q = 0; q < TOPK; ++q) m = fmaxf(m, chval[q] * (1.0f / (float)DIM));
    float s = 0.f, e[TOPK];
    for (int q = 0; q < TOPK; ++q) { e[q] = expf(chval[q] * (1.0f / (float)DIM) - m); s += e[q]; }
    for (int q = 0; q < TOPK; ++q) { wts[b * 8 + q] = e[q] / s; idxs[b * 8 + q] = chosen[q]; }
  }
}

// ---------------------------------------------------------------------------
// Kernel 5: ac_out combine (circular rolls) + residual + LayerNorm1 -> H.
// ---------------------------------------------------------------------------
__global__ __launch_bounds__(256) void ac_ln_kernel(
    const float* __restrict__ S1, const float* __restrict__ wts, const int* __restrict__ idxs,
    const float* __restrict__ lnw, const float* __restrict__ lnb, float* __restrict__ H) {
  const int L = SEQL, D = DIM;
  int b = blockIdx.y, l = blockIdx.x;
  __shared__ float red[256];
  __shared__ float wv[TOPK]; __shared__ int iv[TOPK];
  if (threadIdx.x < TOPK) { wv[threadIdx.x] = wts[b * 8 + threadIdx.x]; iv[threadIdx.x] = idxs[b * 8 + threadIdx.x]; }
  __syncthreads();
  const float* Sb = S1 + (size_t)b * L * D;
  float y[2]; float lsum = 0.f;
  for (int dd = 0; dd < 2; ++dd) {
    int d = threadIdx.x + dd * 256;
    float v = Sb[(size_t)l * D + d];
#pragma unroll
    for (int q = 0; q < TOPK; ++q) {
      int lr = (l - iv[q]) & (L - 1);
      v += wv[q] * Sb[(size_t)lr * D + d];
    }
    y[dd] = v; lsum += v;
  }
  red[threadIdx.x] = lsum; __syncthreads();
  for (int s = 128; s > 0; s >>= 1) { if (threadIdx.x < s) red[threadIdx.x] += red[threadIdx.x + s]; __syncthreads(); }
  float mu = red[0] * (1.0f / (float)D);
  __syncthreads();
  float vs = 0.f;
  for (int dd = 0; dd < 2; ++dd) { float dv = y[dd] - mu; vs += dv * dv; }
  red[threadIdx.x] = vs; __syncthreads();
  for (int s = 128; s > 0; s >>= 1) { if (threadIdx.x < s) red[threadIdx.x] += red[threadIdx.x + s]; __syncthreads(); }
  float rstd = rsqrtf(red[0] * (1.0f / (float)D) + 1e-5f);
  for (int dd = 0; dd < 2; ++dd) {
    int d = threadIdx.x + dd * 256;
    H[(size_t)b * L * D + (size_t)l * D + d] = (y[dd] - mu) * rstd * lnw[d] + lnb[d];
  }
}

// ---------------------------------------------------------------------------
// Kernel 6: split fp32 -> bf16 hi/lo pair (for weights).
// ---------------------------------------------------------------------------
__global__ void split_bf16_kernel(const float* __restrict__ src,
                                  unsigned short* __restrict__ hi,
                                  unsigned short* __restrict__ lo, int n) {
  int i = blockIdx.x * 256 + threadIdx.x;
  if (i < n) {
    float x = src[i];
    __bf16 h = (__bf16)x;
    __bf16 l = (__bf16)(x - (float)h);
    ((__bf16*)hi)[i] = h;
    ((__bf16*)lo)[i] = l;
  }
}

// ---------------------------------------------------------------------------
// Kernel 7: GEMM  C[M x 512] = act(A[M x 512] @ W[512 x 512] + bias)
// A fp32 split to bf16 hi/lo in LDS; W tiles DMA'd by the TDM.
// 3 WMMAs per k-step emulate fp32: Ah*Wh + Ah*Wl + Al*Wh.
// grid: (M/64, 8), block 128 (4 waves: 16 rows each, 4 n-subtiles).
// ---------------------------------------------------------------------------
__global__ __launch_bounds__(128) void gemm_bf16_split_kernel(
    const float* __restrict__ A, const unsigned short* __restrict__ Whi_,
    const unsigned short* __restrict__ Wlo_, const float* __restrict__ bias,
    float* __restrict__ C, int do_gelu) {
  const int D = DIM;
  __shared__ __bf16 Ah[64][36];
  __shared__ __bf16 Al[64][36];
  __shared__ __bf16 Bh[32][64];   // packed: written by TDM
  __shared__ __bf16 Bl[32][64];
  int i0 = blockIdx.x * 64;
  int n0 = blockIdx.y * 64;
  int t = threadIdx.x, lane = t & 31, w = t >> 5;
  v8f acc[4] = {};
  for (int k0 = 0; k0 < D; k0 += 32) {
    // --- async: Tensor Data Mover pulls both 32x64 bf16 weight tiles ---
    if (t < 32) {
      tdm_load_2d_bf16(&Whi_[(size_t)k0 * D + n0],
                       (unsigned)(size_t)(&Bh[0][0]), 64, 32, D);
      tdm_load_2d_bf16(&Wlo_[(size_t)k0 * D + n0],
                       (unsigned)(size_t)(&Bl[0][0]), 64, 32, D);
    }
    // --- meanwhile all threads stage + split the fp32 A tile ---
    for (int e = t; e < 64 * 8; e += 128) {       // 64 rows x 8 float4
      int r = e >> 3, c4 = (e & 7) << 2;
      float4 x4 = *(const float4*)(&A[(size_t)(i0 + r) * D + k0 + c4]);
      float xs[4] = {x4.x, x4.y, x4.z, x4.w};
#pragma unroll
      for (int q = 0; q < 4; ++q) {
        __bf16 h = (__bf16)xs[q];
        Ah[r][c4 + q] = h;
        Al[r][c4 + q] = (__bf16)(xs[q] - (float)h);
      }
    }
    if (k0 + 32 < D)  // prefetch next A k-slab (global_prefetch_b8)
      __builtin_prefetch(&A[(size_t)(i0 + (t >> 1)) * D + k0 + 32 + ((t & 1) << 4)], 0, 0);
    if (t < 32) __builtin_amdgcn_s_wait_tensorcnt(0);
    __syncthreads();
    v16bf ah = load_frag_a(&Ah[w * 16][0], 36, lane);
    v16bf al = load_frag_a(&Al[w * 16][0], 36, lane);
#pragma unroll
    for (int nt = 0; nt < 4; ++nt) {
      v16bf bh = load_frag_b(&Bh[0][0], 64, lane, nt);
      v16bf bl = load_frag_b(&Bl[0][0], 64, lane, nt);
      acc[nt] = wmma_bf16(ah, bh, acc[nt]);
      acc[nt] = wmma_bf16(ah, bl, acc[nt]);
      acc[nt] = wmma_bf16(al, bh, acc[nt]);
    }
    __syncthreads();
  }
  int kg = lane >> 4, colb = lane & 15;
#pragma unroll
  for (int nt = 0; nt < 4; ++nt) {
    int col = n0 + nt * 16 + colb;
    float bval = bias[col];
#pragma unroll
    for (int v = 0; v < 8; ++v) {
      int row = i0 + w * 16 + v + 8 * kg;
      float x = acc[nt][v] + bval;
      if (do_gelu) x = 0.5f * x * (1.0f + erff(x * 0.70710678f));
      C[(size_t)row * D + col] = x;
    }
  }
}

// ---------------------------------------------------------------------------
// Kernel 8: out = LayerNorm2(X + Y). grid: B*L blocks, 256 threads.
// ---------------------------------------------------------------------------
__global__ __launch_bounds__(256) void add_ln_kernel(
    const float* __restrict__ X, const float* __restrict__ Y,
    const float* __restrict__ lnw, const float* __restrict__ lnb, float* __restrict__ Out) {
  const int D = DIM;
  size_t row = blockIdx.x;
  __shared__ float red[256];
  float y[2]; float lsum = 0.f;
  for (int dd = 0; dd < 2; ++dd) {
    int d = threadIdx.x + dd * 256;
    float v = X[row * D + d] + Y[row * D + d];
    y[dd] = v; lsum += v;
  }
  red[threadIdx.x] = lsum; __syncthreads();
  for (int s = 128; s > 0; s >>= 1) { if (threadIdx.x < s) red[threadIdx.x] += red[threadIdx.x + s]; __syncthreads(); }
  float mu = red[0] * (1.0f / (float)D);
  __syncthreads();
  float vs = 0.f;
  for (int dd = 0; dd < 2; ++dd) { float dv = y[dd] - mu; vs += dv * dv; }
  red[threadIdx.x] = vs; __syncthreads();
  for (int s = 128; s > 0; s >>= 1) { if (threadIdx.x < s) red[threadIdx.x] += red[threadIdx.x + s]; __syncthreads(); }
  float rstd = rsqrtf(red[0] * (1.0f / (float)D) + 1e-5f);
  for (int dd = 0; dd < 2; ++dd) {
    int d = threadIdx.x + dd * 256;
    Out[row * D + d] = (y[dd] - mu) * rstd * lnw[d] + lnb[d];
  }
}

// ---------------------------------------------------------------------------
extern "C" void kernel_launch(void* const* d_in, const int* in_sizes, int n_in,
                              void* d_out, int out_size, void* d_ws, size_t ws_size,
                              hipStream_t stream) {
  (void)in_sizes; (void)n_in; (void)out_size; (void)ws_size;
  const float* x    = (const float*)d_in[0];
  const float* w1   = (const float*)d_in[1];
  const float* b1   = (const float*)d_in[2];
  const float* w2   = (const float*)d_in[3];
  const float* b2   = (const float*)d_in[4];
  const float* ln1w = (const float*)d_in[5];
  const float* ln1b = (const float*)d_in[6];
  const float* ln2w = (const float*)d_in[7];
  const float* ln2b = (const float*)d_in[8];
  float* out = (float*)d_out;

  const size_t NE = (size_t)BATCH * SEQL * DIM;   // 33,554,432 elements
  char* p = (char*)d_ws;
  float* s1 = (float*)p; p += NE * 4;
  float* t1 = (float*)p; p += NE * 4;             // later reused as g1
  float* hb = (float*)p; p += NE * 4;             // later reused as g2
  float* s2 = (float*)p; p += NE * 4;
  float* scores = (float*)p; p += (size_t)BATCH * SEQL * 4;
  float* wts = (float*)p; p += BATCH * 8 * 4;
  int*   idxs = (int*)p; p += BATCH * 8 * 4;
  unsigned short* w1hi = (unsigned short*)p; p += (size_t)DIM * DIM * 2;
  unsigned short* w1lo = (unsigned short*)p; p += (size_t)DIM * DIM * 2;
  unsigned short* w2hi = (unsigned short*)p; p += (size_t)DIM * DIM * 2;
  unsigned short* w2lo = (unsigned short*)p; p += (size_t)DIM * DIM * 2;

  float* seasonal_out = out;        // first NE floats
  float* trend_out    = out + NE;   // second NE floats

  // weight splits (independent, launch first)
  split_bf16_kernel<<<dim3((DIM * DIM + 255) / 256), dim3(256), 0, stream>>>(w1, w1hi, w1lo, DIM * DIM);
  split_bf16_kernel<<<dim3((DIM * DIM + 255) / 256), dim3(256), 0, stream>>>(w2, w2hi, w2lo, DIM * DIM);

  // 1) decomp1: x -> s1, t1
  decomp_kernel<<<dim3(SEQL / 128, BATCH), dim3(256), 0, stream>>>(x, s1, t1, (const float*)nullptr);

  // 2) autocorrelation scores via WMMA Gram
  zero_kernel<<<dim3((BATCH * SEQL + 255) / 256), dim3(256), 0, stream>>>(scores, BATCH * SEQL);
  gram_scores_kernel<<<dim3(SEQL / 64, SEQL / 64, BATCH), dim3(128), 0, stream>>>(s1, scores);

  // 3) top-5 + softmax
  topk_softmax_kernel<<<dim3(BATCH), dim3(256), 0, stream>>>(scores, wts, idxs);

  // 4) roll-combine + residual + LN1 -> hb
  ac_ln_kernel<<<dim3(SEQL, BATCH), dim3(256), 0, stream>>>(s1, wts, idxs, ln1w, ln1b, hb);

  // 5) decomp2: hb -> s2 ; trend_comp = t1 + trend2 -> out directly
  decomp_kernel<<<dim3(SEQL / 128, BATCH), dim3(256), 0, stream>>>(hb, s2, trend_out, t1);

  // 6) FFN GEMM1 (+GELU) into g1 (reuses t1), GEMM2 into g2 (reuses hb)
  float* g1 = t1;
  float* g2 = hb;
  const int M = BATCH * SEQL;
  gemm_bf16_split_kernel<<<dim3(M / 64, DIM / 64), dim3(128), 0, stream>>>(s2, w1hi, w1lo, b1, g1, 1);
  gemm_bf16_split_kernel<<<dim3(M / 64, DIM / 64), dim3(128), 0, stream>>>(g1, w2hi, w2lo, b2, g2, 0);

  // 7) seasonal_out = LN2(s2 + g2)
  add_ln_kernel<<<dim3(M), dim3(256), 0, stream>>>(s2, g2, ln2w, ln2b, seasonal_out);
}
